// GCN_60455959658959
// MI455X (gfx1250) — matrix operand order repaired and checked by reading, not compiled, not run on hardware
//
#include <hip/hip_runtime.h>
#include <hip/hip_bf16.h>

// ---------------------------------------------------------------------------
// GCN with learned adjacency for MI455X (gfx1250), wave32 + WMMA f16.
//
// Pipeline:
//  K0 prep   : n1,n2 (40x40 MLP, fp32 VALU) -> P=[n1,-n2,0], Q=[n2,n1,0] f16 [N,128]
//  K1 wtrans : W1^T [128][64] f16, W2^T [32][128] f16
//  K2 sgemm  : a = P Q^T (WMMA f16, K=128) -> St[j][i] = (a[i,j]>0) + (i==j)  (f16 0/1)
//  K3 dinv   : dinv[j] = rsqrt(row-sum of St[j][:])
//  K4 xw1    : Gt1[(b,f)][i] = dinv[i] * (x W1)[b,i,f]          (WMMA, K=64)
//  K5 gemm1  : H1r[(b,j)][f] = relu(dinv[j]*Sum_i St[j][i]*Gt1[(b,f)][i] + b1[f])
//              (64x64 per wave: 16 WMMAs per K-step, 1 b128 load per WMMA)
//  K6 hw2    : G2t[(b,f2)][j] = dinv[j] * (H1 W2)[b,j,f2]       (WMMA, K=128)
//  K7 gemm2  : out = softmax(dinv[j2]*Sum_j St[j2][j]*G2t[(b,f2)][j] + b2[f2])
// ---------------------------------------------------------------------------

#define N_NODES 4096
#define BATCH   32
#define DIN     64
#define DH      128
#define DOUT    32
#define DIME    40

typedef __attribute__((ext_vector_type(16))) _Float16 v16h;
typedef __attribute__((ext_vector_type(8)))  _Float16 v8h;
typedef __attribute__((ext_vector_type(8)))  float    v8f;

union F16x16 { v16h v; v8h h[2]; };

// A-fragment (16x32 f16, M x K): lane<16 -> K {k0..k0+7, k0+16..k0+23},
// lane>=16 -> K {k0+8..k0+15, k0+24..k0+31}.  Row-major source [rows][stride].
__device__ __forceinline__ v16h load_a_frag(const _Float16* base, int stride,
                                            int m0, int k0) {
    int lane = threadIdx.x & 31;
    int hi = lane >> 4;
    const _Float16* p = base + (size_t)(m0 + (lane & 15)) * stride + k0 + hi * 8;
    F16x16 u;
    u.h[0] = *(const v8h*)(p);
    u.h[1] = *(const v8h*)(p + 16);
    return u.v;
}

// B-fragment (32x16 f16, K x N) loaded from a ROW-MAJOR TRANSPOSED array
// Bt[n][k]: lane<16 -> K {k0..k0+7, k0+16..k0+23}, lane>=16 -> K {k0+8..k0+15, k0+24..k0+31}.
__device__ __forceinline__ v16h load_b_frag(const _Float16* baseT, int stride,
                                            int n0, int k0) {
    int lane = threadIdx.x & 31;
    int hi = lane >> 4;
    const _Float16* p = baseT + (size_t)(n0 + (lane & 15)) * stride + k0 + hi * 8;
    F16x16 u;
    u.h[0] = *(const v8h*)(p);
    u.h[1] = *(const v8h*)(p + 16);
    return u.v;
}

// A-fragment directly from an f32 row-major source (converted to f16).
__device__ __forceinline__ v16h load_a_frag_f32(const float* base, int stride,
                                                int m0, int k0) {
    int lane = threadIdx.x & 31;
    int hi = lane >> 4;
    const float* p = base + (size_t)(m0 + (lane & 15)) * stride + k0 + hi * 8;
    v16h v;
#pragma unroll
    for (int e = 0; e < 8; ++e) v[e] = (_Float16)p[e];
#pragma unroll
    for (int e = 0; e < 8; ++e) v[8 + e] = (_Float16)p[16 + e];
    return v;
}

// Per-lane row prefetch (emits global_prefetch_b8); one line per fragment row.
__device__ __forceinline__ void prefetch_row(const _Float16* base, int stride,
                                             int r0, int k) {
    int lane = threadIdx.x & 31;
    const _Float16* p = base + (size_t)(r0 + (lane & 15)) * stride + k;
    __builtin_prefetch((const void*)p, 0, 1);
}

__device__ __forceinline__ v8f wmma_f16(v16h a, v16h b, v8f c) {
    return __builtin_amdgcn_wmma_f32_16x16x32_f16(false, a, false, b,
                                                  (short)0, c, false, false);
}

// --------------------------- K0: embedding MLP -----------------------------
__global__ void prep_kernel(const float* __restrict__ emb1,
                            const float* __restrict__ emb2,
                            const float* __restrict__ l1w,
                            const float* __restrict__ l1b,
                            const float* __restrict__ l2w,
                            const float* __restrict__ l2b,
                            _Float16* __restrict__ P, _Float16* __restrict__ Q) {
    int i = blockIdx.x * blockDim.x + threadIdx.x;
    if (i >= N_NODES) return;
    float e1[DIME], e2[DIME];
#pragma unroll 8
    for (int k = 0; k < DIME; ++k) {
        e1[k] = emb1[i * DIME + k];
        e2[k] = emb2[i * DIME + k];
    }
    _Float16* prow = P + (size_t)i * 128;
    _Float16* qrow = Q + (size_t)i * 128;
    for (int d = 0; d < DIME; ++d) {
        float s1 = l1b[d], s2 = l2b[d];
#pragma unroll 8
        for (int k = 0; k < DIME; ++k) {
            s1 += e1[k] * l1w[k * DIME + d];
            s2 += e2[k] * l2w[k * DIME + d];
        }
        float n1 = tanhf(3.0f * s1);
        float n2 = tanhf(3.0f * s2);
        prow[d]        = (_Float16)n1;
        prow[DIME + d] = (_Float16)(-n2);
        qrow[d]        = (_Float16)n2;
        qrow[DIME + d] = (_Float16)n1;
    }
    for (int d = 2 * DIME; d < 128; ++d) {
        prow[d] = (_Float16)0.f;
        qrow[d] = (_Float16)0.f;
    }
}

// --------------------------- K1: weight transposes -------------------------
__global__ void wtrans_kernel(const float* __restrict__ w1,
                              const float* __restrict__ w2,
                              _Float16* __restrict__ w1t,
                              _Float16* __restrict__ w2t) {
    for (int idx = threadIdx.x; idx < DIN * DH; idx += blockDim.x) {
        int d = idx >> 7, f = idx & (DH - 1);            // w1[d][f]
        w1t[f * DIN + d] = (_Float16)w1[idx];
    }
    for (int idx = threadIdx.x; idx < DH * DOUT; idx += blockDim.x) {
        int f = idx >> 5, f2 = idx & (DOUT - 1);         // w2[f][f2]
        w2t[f2 * DH + f] = (_Float16)w2[idx];
    }
}

// --------------------------- K2: sign GEMM ---------------------------------
// a[i,j] = sum_k P[i,k]*Q[j,k];  St[j][i] = (a>0) | (i==j), stored f16 0/1.
__global__ void sign_gemm_kernel(const _Float16* __restrict__ P,
                                 const _Float16* __restrict__ Q,
                                 _Float16* __restrict__ St) {
    int wave = threadIdx.x >> 5;
    int lane = threadIdx.x & 31;
    int i0 = blockIdx.x * 16;                  // M tile (i)
    int j0 = (blockIdx.y * 8 + wave) * 16;     // N tile (j)
    v16h a[4], b[4];
#pragma unroll
    for (int k = 0; k < 4; ++k) a[k] = load_a_frag(P, 128, i0, k * 32);
#pragma unroll
    for (int k = 0; k < 4; ++k) b[k] = load_b_frag(Q, 128, j0, k * 32);
    v8f acc = {};
#pragma unroll
    for (int k = 0; k < 4; ++k) acc = wmma_f16(a[k], b[k], acc);

    int n = lane & 15, hi = lane >> 4;
    int j = j0 + n;
    int ibase = i0 + hi * 8;
    v8h out;
#pragma unroll
    for (int r = 0; r < 8; ++r) {
        int i = ibase + r;
        float s = (acc[r] > 0.f) ? 1.f : 0.f;
        if (i == j) s = 1.f;                   // self-loop (diag of a is 0)
        out[r] = (_Float16)s;
    }
    *(v8h*)(St + (size_t)j * N_NODES + ibase) = out;
}

// --------------------------- K3: degree -> dinv ----------------------------
__global__ void dinv_kernel(const _Float16* __restrict__ St,
                            float* __restrict__ dinv) {
    int j = blockIdx.x;
    const v8h* row = (const v8h*)(St + (size_t)j * N_NODES);
    float s = 0.f;
    for (int v = threadIdx.x; v < N_NODES / 8; v += blockDim.x) {
        v8h d = row[v];
#pragma unroll
        for (int e = 0; e < 8; ++e) s += (float)d[e];
    }
#pragma unroll
    for (int m = 16; m >= 1; m >>= 1) s += __shfl_xor(s, m, 32);
    __shared__ float red[8];
    int wave = threadIdx.x >> 5, lane = threadIdx.x & 31;
    if (lane == 0) red[wave] = s;
    __syncthreads();
    if (threadIdx.x == 0) {
        float t = 0.f;
        for (int w = 0; w < 8; ++w) t += red[w];
        dinv[j] = rsqrtf(t);
    }
}

// --------------------------- K4: Gt1 = dinv * (X W1), transposed -----------
__global__ void xw1_kernel(const float* __restrict__ x,
                           const _Float16* __restrict__ w1t,
                           const float* __restrict__ dinv,
                           _Float16* __restrict__ Gt1) {
    int wave = threadIdx.x >> 5;               // n-tile (f) 0..7
    int lane = threadIdx.x & 31;
    int m0 = blockIdx.x * 16;                  // rows of x: m = b*N + i
    int n0 = wave * 16;
    v8f acc = {};
#pragma unroll
    for (int k = 0; k < 2; ++k) {
        v16h a = load_a_frag_f32(x, DIN, m0, k * 32);
        v16h b = load_b_frag(w1t, DIN, n0, k * 32);
        acc = wmma_f16(a, b, acc);
    }
    int n = lane & 15, hi = lane >> 4;
    int f = n0 + n;
    int mb = m0 + hi * 8;
    int bb = mb >> 12;                         // batch
    int i0 = mb & (N_NODES - 1);
    v8h out;
#pragma unroll
    for (int r = 0; r < 8; ++r)
        out[r] = (_Float16)(acc[r] * dinv[i0 + r]);
    *(v8h*)(Gt1 + ((size_t)bb * DH + f) * N_NODES + i0) = out;
}

// --------------------------- K5: main propagation GEMM ---------------------
// C[j,(b,f)] = sum_i St[j][i] * Gt1[(b,f)][i];  H1r = relu(dinv[j]*C + b1[f])
// 64x64 output per wave: 4x4 tiles, 16 WMMAs per 32-wide K-step.
__global__ void gemm1_kernel(const _Float16* __restrict__ St,
                             const _Float16* __restrict__ Gt1,
                             const float* __restrict__ dinv,
                             const float* __restrict__ bias1,
                             _Float16* __restrict__ H1r) {
    int wave = threadIdx.x >> 5;
    int lane = threadIdx.x & 31;
    int m0 = blockIdx.x * 64;                   // j panel (shared by 8 waves)
    int n0 = (blockIdx.y * 8 + wave) * 64;      // c = b*128+f panel
    v8f acc[4][4] = {};
    for (int kk = 0; kk < N_NODES / 32; ++kk) {
        int k0 = kk * 32;
        if ((kk & 7) == 0 && kk < 120) {        // wave-uniform: EXEC stays all-1
            int kp = k0 + 512;                  // 1KB lookahead per row
#pragma unroll
            for (int t = 0; t < 4; ++t) {
                prefetch_row(St,  N_NODES, m0 + t * 16, kp);
                prefetch_row(Gt1, N_NODES, n0 + t * 16, kp);
            }
        }
        v16h a[4], b[4];
#pragma unroll
        for (int t = 0; t < 4; ++t) a[t] = load_a_frag(St,  N_NODES, m0 + t * 16, k0);
#pragma unroll
        for (int t = 0; t < 4; ++t) b[t] = load_b_frag(Gt1, N_NODES, n0 + t * 16, k0);
#pragma unroll
        for (int ti = 0; ti < 4; ++ti)
#pragma unroll
            for (int tj = 0; tj < 4; ++tj)
                acc[ti][tj] = wmma_f16(a[ti], b[tj], acc[ti][tj]);
    }
    int n = lane & 15, hi = lane >> 4;
#pragma unroll
    for (int ti = 0; ti < 4; ++ti)
#pragma unroll
        for (int tj = 0; tj < 4; ++tj) {
            int c = n0 + tj * 16 + n;
            int bb = c >> 7, f = c & (DH - 1);
            float bias = bias1[f];
#pragma unroll
            for (int r = 0; r < 8; ++r) {
                int j = m0 + ti * 16 + hi * 8 + r;
                float h = fmaxf(dinv[j] * acc[ti][tj][r] + bias, 0.f);
                H1r[((size_t)bb * N_NODES + j) * DH + f] = (_Float16)h;
            }
        }
}

// --------------------------- K6: G2t = dinv * (H1 W2), transposed ----------
__global__ void hw2_kernel(const _Float16* __restrict__ H1r,
                           const _Float16* __restrict__ w2t,
                           const float* __restrict__ dinv,
                           _Float16* __restrict__ G2t) {
    int wave = threadIdx.x >> 5;               // n-tile (f2) 0/1
    int lane = threadIdx.x & 31;
    int m0 = blockIdx.x * 16;                  // m = b*N + j
    int n0 = wave * 16;
    v8f acc = {};
#pragma unroll
    for (int k = 0; k < 4; ++k) {
        v16h a = load_a_frag(H1r, DH, m0, k * 32);
        v16h b = load_b_frag(w2t, DH, n0, k * 32);
        acc = wmma_f16(a, b, acc);
    }
    int n = lane & 15, hi = lane >> 4;
    int f2 = n0 + n;
    int mb = m0 + hi * 8;
    int bb = mb >> 12;
    int j0 = mb & (N_NODES - 1);
    v8h out;
#pragma unroll
    for (int r = 0; r < 8; ++r)
        out[r] = (_Float16)(acc[r] * dinv[j0 + r]);
    *(v8h*)(G2t + ((size_t)bb * DOUT + f2) * N_NODES + j0) = out;
}

// --------------------------- K7: layer-2 GEMM + bias + softmax -------------
// 32 rows (j2) x 32 cols (f2 of one batch) per wave; fused softmax epilogue.
__global__ void gemm2_kernel(const _Float16* __restrict__ St,
                             const _Float16* __restrict__ G2t,
                             const float* __restrict__ dinv,
                             const float* __restrict__ bias2,
                             float* __restrict__ out) {
    int wave = threadIdx.x >> 5;
    int lane = threadIdx.x & 31;
    int m0 = blockIdx.x * 32;                  // j2 (two M tiles)
    int bb = blockIdx.y * 8 + wave;            // batch 0..31
    int c0 = bb * DOUT;                        // 32 columns -> 2 n-tiles
    v8f acc[2][2] = {};
    for (int kk = 0; kk < N_NODES / 32; ++kk) {
        int k0 = kk * 32;
        v16h a0  = load_a_frag(St,  N_NODES, m0,      k0);
        v16h a1  = load_a_frag(St,  N_NODES, m0 + 16, k0);
        v16h bf0 = load_b_frag(G2t, N_NODES, c0,      k0);
        v16h bf1 = load_b_frag(G2t, N_NODES, c0 + 16, k0);
        acc[0][0] = wmma_f16(a0, bf0, acc[0][0]);
        acc[0][1] = wmma_f16(a0, bf1, acc[0][1]);
        acc[1][0] = wmma_f16(a1, bf0, acc[1][0]);
        acc[1][1] = wmma_f16(a1, bf1, acc[1][1]);
    }
    int n = lane & 15, hi = lane >> 4;
    float bias0 = bias2[n], bias1 = bias2[16 + n];
#pragma unroll
    for (int ti = 0; ti < 2; ++ti) {
#pragma unroll
        for (int r = 0; r < 8; ++r) {
            int j2 = m0 + ti * 16 + hi * 8 + r;
            float dj = dinv[j2];
            float v0 = dj * acc[ti][0][r] + bias0;   // f2 = n
            float v1 = dj * acc[ti][1][r] + bias1;   // f2 = 16 + n
            // softmax over 32 f2 values, spread across 16 lanes x 2 regs;
            // xor masks < 16 keep exchanges inside each half-wave (same j2).
            float mx = fmaxf(v0, v1);
#pragma unroll
            for (int msk = 1; msk < 16; msk <<= 1)
                mx = fmaxf(mx, __shfl_xor(mx, msk, 32));
            float e0 = __expf(v0 - mx);
            float e1 = __expf(v1 - mx);
            float s = e0 + e1;
#pragma unroll
            for (int msk = 1; msk < 16; msk <<= 1)
                s += __shfl_xor(s, msk, 32);
            float inv = 1.f / s;
            size_t row = (size_t)bb * N_NODES + j2;
            out[row * DOUT + n]      = e0 * inv;
            out[row * DOUT + 16 + n] = e1 * inv;
        }
    }
}

// ---------------------------------------------------------------------------
extern "C" void kernel_launch(void* const* d_in, const int* in_sizes, int n_in,
                              void* d_out, int out_size, void* d_ws, size_t ws_size,
                              hipStream_t stream) {
    const float* x     = (const float*)d_in[0];
    const float* emb1  = (const float*)d_in[1];
    const float* emb2  = (const float*)d_in[2];
    const float* l1w   = (const float*)d_in[3];
    const float* l1b   = (const float*)d_in[4];
    const float* l2w   = (const float*)d_in[5];
    const float* l2b   = (const float*)d_in[6];
    const float* g1w   = (const float*)d_in[7];
    const float* g1b   = (const float*)d_in[8];
    const float* g2w   = (const float*)d_in[9];
    const float* g2b   = (const float*)d_in[10];
    float* out = (float*)d_out;

    char* ws = (char*)d_ws;
    size_t oP    = 0;
    size_t oQ    = oP  + (size_t)N_NODES * 128 * 2;                 // 1 MB
    size_t oSt   = oQ  + (size_t)N_NODES * 128 * 2;                 // 1 MB
    size_t oGt1  = oSt + (size_t)N_NODES * N_NODES * 2;             // 32 MB
    size_t oH1   = oGt1 + (size_t)BATCH * DH * N_NODES * 2;         // 32 MB
    size_t oG2   = oH1  + (size_t)BATCH * N_NODES * DH * 2;         // 32 MB
    size_t oDinv = oG2  + (size_t)BATCH * DOUT * N_NODES * 2;       // 8 MB
    size_t oW1t  = oDinv + (size_t)N_NODES * 4;
    size_t oW2t  = oW1t  + (size_t)DH * DIN * 2;

    _Float16* P    = (_Float16*)(ws + oP);
    _Float16* Q    = (_Float16*)(ws + oQ);
    _Float16* St   = (_Float16*)(ws + oSt);
    _Float16* Gt1  = (_Float16*)(ws + oGt1);
    _Float16* H1r  = (_Float16*)(ws + oH1);
    _Float16* G2t  = (_Float16*)(ws + oG2);
    float*    dinv = (float*)(ws + oDinv);
    _Float16* w1t  = (_Float16*)(ws + oW1t);
    _Float16* w2t  = (_Float16*)(ws + oW2t);

    prep_kernel<<<dim3(N_NODES / 256), dim3(256), 0, stream>>>(
        emb1, emb2, l1w, l1b, l2w, l2b, P, Q);
    wtrans_kernel<<<dim3(1), dim3(256), 0, stream>>>(g1w, g2w, w1t, w2t);
    sign_gemm_kernel<<<dim3(N_NODES / 16, 32), dim3(256), 0, stream>>>(P, Q, St);
    dinv_kernel<<<dim3(N_NODES), dim3(256), 0, stream>>>(St, dinv);
    xw1_kernel<<<dim3(BATCH * N_NODES / 16), dim3(256), 0, stream>>>(
        x, w1t, dinv, Gt1);
    gemm1_kernel<<<dim3(N_NODES / 64, 8), dim3(256), 0, stream>>>(
        St, Gt1, dinv, g1b, H1r);
    hw2_kernel<<<dim3(BATCH * N_NODES / 16), dim3(64), 0, stream>>>(
        H1r, w2t, dinv, G2t);
    gemm2_kernel<<<dim3(N_NODES / 32, 4), dim3(256), 0, stream>>>(
        St, G2t, dinv, g2b, out);
}
